// WindowTransformer_22428319220220
// MI455X (gfx1250) — matrix operand order repaired and checked by reading, not compiled, not run on hardware
//
#include <hip/hip_runtime.h>
#include <hip/hip_bf16.h>
#include <math.h>

// ---------------- types ----------------
typedef _Float16 f16;
typedef __attribute__((ext_vector_type(16))) _Float16 v16h;
typedef __attribute__((ext_vector_type(8)))  _Float16 v8h;
typedef __attribute__((ext_vector_type(8)))  float    v8f;

union Frag16 { v16h v; v8h h[2]; };

// ---------------- problem constants ----------------
#define L_LAYERS 2
#define DMODEL   768
#define DFF      3072
#define NHEADS   12
#define HEADD    64
#define NWIN     128
#define NTOK     256
#define MTOT     32768
#define EPSV     1e-6f

// GEMM tiling: block tile 128x128, BK=32, 8 waves as 2(M) x 4(N), wave tile 64x32
#define BM 128
#define BN 128
#define BK 32
#define LDSPAD 40                      // BK + 8 halfs (80B rows, 16B-multiple)
#define TILE_HALFS (128 * LDSPAD)      // one matrix, one stage: 5120 halfs
#define STAGE_HALFS (2 * TILE_HALFS)   // A + B per stage

enum GemmMode { MODE_F16 = 0, MODE_GELU = 2, MODE_RES = 3 };

// attention LDS layout (halfs)
#define QS_STRIDE 72
#define SS_STRIDE 264
#define SMEM_Q   0
#define SMEM_K   (256*QS_STRIDE)
#define SMEM_VT  (2*256*QS_STRIDE)
#define SMEM_S   (2*256*QS_STRIDE + 64*SS_STRIDE)
#define SMEM_ATTN_BYTES ((2*256*QS_STRIDE + 64*SS_STRIDE + 256*SS_STRIDE) * 2)

// ---------------- CDNA5 async global->LDS helpers ----------------
// Flat LDS addresses carry the wave-relative LDS byte offset in addr[31:0]
// (ISA 10.2 aperture rule), which is exactly what the async VDST VGPR wants.
__device__ __forceinline__ uint32_t lds_off(const void* p) {
  return (uint32_t)(uintptr_t)p;
}
__device__ __forceinline__ void async_copy_b128(uint32_t lds, const void* g) {
  asm volatile("global_load_async_to_lds_b128 %0, %1, off"
               :: "v"(lds), "v"((unsigned long long)(uintptr_t)g) : "memory");
}
__device__ __forceinline__ void wait_async_le4() {
  asm volatile("s_wait_asynccnt 0x4" ::: "memory");
}
__device__ __forceinline__ void wait_async_0() {
  asm volatile("s_wait_asynccnt 0x0" ::: "memory");
}

// ---------------- window split / merge ----------------
__global__ __launch_bounds__(256) void win_split(const float* __restrict__ x,
                                                 float* __restrict__ xw) {
  int idx = blockIdx.x * 256 + threadIdx.x;
  int row = idx / 192;
  int c   = (idx % 192) * 4;
  int s  = row >> 8, t = row & 255;
  int hh = t >> 4,  ww = t & 15;
  int win = s >> 3, b = s & 7;
  int i = win >> 2, j = win & 3;
  int rin = b * 4096 + (i * 16 + hh) * 64 + (j * 16 + ww);
  *(float4*)(xw + (size_t)row * DMODEL + c) =
      *(const float4*)(x + (size_t)rin * DMODEL + c);
}

__global__ __launch_bounds__(256) void win_merge(const float* __restrict__ xw,
                                                 float* __restrict__ out) {
  int idx = blockIdx.x * 256 + threadIdx.x;
  int row = idx / 192;
  int c   = (idx % 192) * 4;
  int b = row >> 12, p = row & 4095;
  int hp = p >> 6, wp = p & 63;
  int i = hp >> 4, hh = hp & 15, j = wp >> 4, ww = wp & 15;
  int s  = (i * 4 + j) * 8 + b;
  int ro = s * 256 + hh * 16 + ww;
  *(float4*)(out + (size_t)row * DMODEL + c) =
      *(const float4*)(xw + (size_t)ro * DMODEL + c);
}

// ---------------- weight transpose + f32->f16 ----------------
__global__ __launch_bounds__(256) void transpose_cvt(const float* __restrict__ src,
                                                     f16* __restrict__ dst,
                                                     int R, int C) {
  __shared__ float tile[32][33];
  int ct = blockIdx.x * 32, rt = blockIdx.y * 32;
  int tx = threadIdx.x & 31, ty = threadIdx.x >> 5;
#pragma unroll
  for (int kk = 0; kk < 4; ++kk)
    tile[ty + 8 * kk][tx] = src[(size_t)(rt + ty + 8 * kk) * C + ct + tx];
  __syncthreads();
#pragma unroll
  for (int kk = 0; kk < 4; ++kk)
    dst[(size_t)(ct + ty + 8 * kk) * R + rt + tx] = (f16)tile[tx][ty + 8 * kk];
}

// ---------------- layernorm ----------------
__global__ __launch_bounds__(256) void ln_f16(const float* __restrict__ xw,
                                              const float* __restrict__ scale,
                                              const float* __restrict__ bias,
                                              f16* __restrict__ out) {
  int row = blockIdx.x, t = threadIdx.x;
  const float* xr = xw + (size_t)row * DMODEL;
  float a0 = xr[t], a1 = xr[t + 256], a2 = xr[t + 512];
  __shared__ float red[256];
  red[t] = a0 + a1 + a2;
  __syncthreads();
  for (int o = 128; o > 0; o >>= 1) { if (t < o) red[t] += red[t + o]; __syncthreads(); }
  float mean = red[0] * (1.0f / DMODEL);
  __syncthreads();
  float d0 = a0 - mean, d1 = a1 - mean, d2 = a2 - mean;
  red[t] = d0 * d0 + d1 * d1 + d2 * d2;
  __syncthreads();
  for (int o = 128; o > 0; o >>= 1) { if (t < o) red[t] += red[t + o]; __syncthreads(); }
  float rstd = rsqrtf(red[0] * (1.0f / DMODEL) + EPSV);
  f16* orow = out + (size_t)row * DMODEL;
  orow[t]       = (f16)(d0 * rstd * scale[t]       + bias[t]);
  orow[t + 256] = (f16)(d1 * rstd * scale[t + 256] + bias[t + 256]);
  orow[t + 512] = (f16)(d2 * rstd * scale[t + 512] + bias[t + 512]);
}

// ---------------- WMMA GEMM: async-LDS double-buffered pipeline ----------------
// C[M,N] = A[M,K](f16) * Bt[N,K](f16)^T  with fused epilogues.
__global__ __launch_bounds__(256) void gemm_wmma(const f16* __restrict__ A, int lda,
                                                 const f16* __restrict__ Bt, int ldb,
                                                 const float* __restrict__ bias,
                                                 f16* __restrict__ outh,
                                                 float* __restrict__ outf, int ldo,
                                                 int Kdim, int mode, float scale) {
  __shared__ f16 smem[2 * STAGE_HALFS];     // [stage][A(5120) | B(5120)] = 40KB
  int tid  = threadIdx.x;
  int lane = tid & 31, wave = tid >> 5;
  int wm = wave >> 2, wn = wave & 3;        // 2 x 4 wave grid
  int mrow0 = blockIdx.x * BM;
  int ncol0 = blockIdx.y * BN;
  int lm = lane & 15;
  int lo = (lane < 16) ? 0 : 8;             // A-frag K split
  int hi = (lane < 16) ? 0 : 16;            // unused; B frags also split K by lo

  // per-thread async chunk coords: 512 16B-chunks per tile, 2 per thread
  int c0 = tid, c1 = tid + 256;
  int ra0 = c0 >> 2, sa0 = (c0 & 3) * 8;    // row, half-offset within 32-half row
  int ra1 = c1 >> 2, sa1 = (c1 & 3) * 8;
  const char* gA0 = (const char*)(A  + (size_t)(mrow0 + ra0) * lda + sa0);
  const char* gA1 = (const char*)(A  + (size_t)(mrow0 + ra1) * lda + sa1);
  const char* gB0 = (const char*)(Bt + (size_t)(ncol0 + ra0) * ldb + sa0);
  const char* gB1 = (const char*)(Bt + (size_t)(ncol0 + ra1) * ldb + sa1);
  uint32_t lA0 = lds_off(smem + ra0 * LDSPAD + sa0);
  uint32_t lA1 = lds_off(smem + ra1 * LDSPAD + sa1);
  uint32_t lB0 = lds_off(smem + TILE_HALFS + ra0 * LDSPAD + sa0);
  uint32_t lB1 = lds_off(smem + TILE_HALFS + ra1 * LDSPAD + sa1);
  const uint32_t stageB = STAGE_HALFS * 2;  // stage stride in bytes

  v8f acc[4][2] = {};
  int nsteps = Kdim / BK;

  // prologue: stage 0 in flight
  {
    async_copy_b128(lA0, gA0); async_copy_b128(lA1, gA1);
    async_copy_b128(lB0, gB0); async_copy_b128(lB1, gB1);
  }
  for (int ks = 0; ks < nsteps; ++ks) {
    int stage = ks & 1;
    if (ks + 1 < nsteps) {               // prefetch next stage, then wait current
      size_t go = (size_t)(ks + 1) * (BK * 2);
      uint32_t so = (uint32_t)((ks + 1) & 1) * stageB;
      async_copy_b128(lA0 + so, gA0 + go); async_copy_b128(lA1 + so, gA1 + go);
      async_copy_b128(lB0 + so, gB0 + go); async_copy_b128(lB1 + so, gB1 + go);
      wait_async_le4();
    } else {
      wait_async_0();
    }
    __syncthreads();                      // all waves' async writes visible

    const f16* As = smem + (size_t)stage * STAGE_HALFS;
    const f16* Bs = As + TILE_HALFS;
    Frag16 b[2];
#pragma unroll
    for (int nt = 0; nt < 2; ++nt) {
      const f16* bp = Bs + (wn * 32 + nt * 16 + lm) * LDSPAD;
      b[nt].h[0] = *(const v8h*)(bp + lo);
      b[nt].h[1] = *(const v8h*)(bp + 16 + lo);
    }
#pragma unroll
    for (int mt = 0; mt < 4; ++mt) {
      const f16* ap = As + (wm * 64 + mt * 16 + lm) * LDSPAD;
      Frag16 a;
      a.h[0] = *(const v8h*)(ap + lo);
      a.h[1] = *(const v8h*)(ap + 16 + lo);
#pragma unroll
      for (int nt = 0; nt < 2; ++nt)
        acc[mt][nt] = __builtin_amdgcn_wmma_f32_16x16x32_f16(
            false, a.v, false, b[nt].v, (short)0, acc[mt][nt], false, false);
    }
    __syncthreads();                      // reads done before buffer reuse
  }

#pragma unroll
  for (int mt = 0; mt < 4; ++mt)
#pragma unroll
    for (int nt = 0; nt < 2; ++nt) {
      int col = ncol0 + wn * 32 + nt * 16 + lm;
      float bv = bias[col];
#pragma unroll
      for (int r = 0; r < 8; ++r) {
        int m = mrow0 + wm * 64 + mt * 16 + ((lane < 16) ? r : 8 + r);
        float val = acc[mt][nt][r] + bv;
        if (mode == MODE_F16) {
          outh[(size_t)m * ldo + col] = (f16)(val * scale);
        } else if (mode == MODE_GELU) {
          float g = val / (1.0f + __expf(-1.702f * val));
          outh[(size_t)m * ldo + col] = (f16)g;
        } else {
          outf[(size_t)m * ldo + col] += val;
        }
      }
    }
}

// ---------------- attention: one block = (window, head) ----------------
__global__ __launch_bounds__(256) void attn_wmma(const f16* __restrict__ q,
                                                 const f16* __restrict__ k,
                                                 const f16* __restrict__ v,
                                                 f16* __restrict__ o) {
  extern __shared__ char smem_raw[];
  f16* Qs  = (f16*)smem_raw + SMEM_Q;
  f16* Ks  = (f16*)smem_raw + SMEM_K;
  f16* Vts = (f16*)smem_raw + SMEM_VT;
  f16* Ss  = (f16*)smem_raw + SMEM_S;
  int w = blockIdx.x / NHEADS, h = blockIdx.x % NHEADS;
  int t = threadIdx.x;
  size_t base = (size_t)w * NTOK;
  const f16* qr = q + (base + t) * DMODEL + h * HEADD;
  const f16* kr = k + (base + t) * DMODEL + h * HEADD;
  const f16* vr = v + (base + t) * DMODEL + h * HEADD;
#pragma unroll
  for (int c = 0; c < 8; ++c) {
    *(v8h*)(Qs + t * QS_STRIDE + c * 8) = *(const v8h*)(qr + c * 8);
    *(v8h*)(Ks + t * QS_STRIDE + c * 8) = *(const v8h*)(kr + c * 8);
    v8h vv = *(const v8h*)(vr + c * 8);
#pragma unroll
    for (int j = 0; j < 8; ++j) Vts[(c * 8 + j) * SS_STRIDE + t] = vv[j];
  }
  __syncthreads();

  int lane = t & 31, wave = t >> 5;
  int lm = lane & 15;
  int lo = (lane < 16) ? 0 : 8;
  int hi = (lane < 16) ? 0 : 16;

  // S = Q * K^T (q pre-scaled by 1/8)
  for (int idx = wave; idx < 256; idx += 8) {
    int mt = idx >> 4, nt = idx & 15;
    v8f acc = {};
#pragma unroll
    for (int ks = 0; ks < 2; ++ks) {
      int k0 = ks * 32;
      Frag16 a, b;
      a.h[0] = *(const v8h*)(Qs + (mt * 16 + lm) * QS_STRIDE + k0 + lo);
      a.h[1] = *(const v8h*)(Qs + (mt * 16 + lm) * QS_STRIDE + k0 + 16 + lo);
      b.h[0] = *(const v8h*)(Ks + (nt * 16 + lm) * QS_STRIDE + k0 + lo);
      b.h[1] = *(const v8h*)(Ks + (nt * 16 + lm) * QS_STRIDE + k0 + 16 + lo);
      acc = __builtin_amdgcn_wmma_f32_16x16x32_f16(
          false, a.v, false, b.v, (short)0, acc, false, false);
    }
#pragma unroll
    for (int r = 0; r < 8; ++r) {
      int m = mt * 16 + ((lane < 16) ? r : 8 + r);
      Ss[m * SS_STRIDE + nt * 16 + lm] = (f16)acc[r];
    }
  }
  __syncthreads();

  // softmax, one thread per row
  {
    f16* sr = Ss + t * SS_STRIDE;
    float mx = -1e30f;
#pragma unroll 4
    for (int c = 0; c < 32; ++c) {
      v8h sv = *(const v8h*)(sr + c * 8);
#pragma unroll
      for (int j = 0; j < 8; ++j) mx = fmaxf(mx, (float)sv[j]);
    }
    float sum = 0.0f;
#pragma unroll 4
    for (int c = 0; c < 32; ++c) {
      v8h sv = *(const v8h*)(sr + c * 8);
      v8h ev;
#pragma unroll
      for (int j = 0; j < 8; ++j) {
        float e = __expf((float)sv[j] - mx);
        sum += e;
        ev[j] = (f16)e;
      }
      *(v8h*)(sr + c * 8) = ev;
    }
    float inv = 1.0f / sum;
#pragma unroll 4
    for (int c = 0; c < 32; ++c) {
      v8h sv = *(const v8h*)(sr + c * 8);
#pragma unroll
      for (int j = 0; j < 8; ++j) sv[j] = (f16)((float)sv[j] * inv);
      *(v8h*)(sr + c * 8) = sv;
    }
  }
  __syncthreads();

  // O = P * V
  v8f acc[2][4] = {};
  for (int kk = 0; kk < 8; ++kk) {
    int k0 = kk * 32;
    Frag16 b[4];
#pragma unroll
    for (int nt = 0; nt < 4; ++nt) {
      b[nt].h[0] = *(const v8h*)(Vts + (nt * 16 + lm) * SS_STRIDE + k0 + lo);
      b[nt].h[1] = *(const v8h*)(Vts + (nt * 16 + lm) * SS_STRIDE + k0 + 16 + lo);
    }
#pragma unroll
    for (int mi = 0; mi < 2; ++mi) {
      int mt = wave * 2 + mi;
      Frag16 a;
      a.h[0] = *(const v8h*)(Ss + (mt * 16 + lm) * SS_STRIDE + k0 + lo);
      a.h[1] = *(const v8h*)(Ss + (mt * 16 + lm) * SS_STRIDE + k0 + 16 + lo);
#pragma unroll
      for (int nt = 0; nt < 4; ++nt)
        acc[mi][nt] = __builtin_amdgcn_wmma_f32_16x16x32_f16(
            false, a.v, false, b[nt].v, (short)0, acc[mi][nt], false, false);
    }
  }
#pragma unroll
  for (int mi = 0; mi < 2; ++mi)
#pragma unroll
    for (int nt = 0; nt < 4; ++nt)
#pragma unroll
      for (int r = 0; r < 8; ++r) {
        int m = wave * 32 + mi * 16 + ((lane < 16) ? r : 8 + r);
        o[(base + m) * DMODEL + h * HEADD + nt * 16 + lm] = (f16)acc[mi][nt][r];
      }
}

// ---------------- host orchestration ----------------
extern "C" void kernel_launch(void* const* d_in, const int* in_sizes, int n_in,
                              void* d_out, int out_size, void* d_ws, size_t ws_size,
                              hipStream_t stream) {
  (void)in_sizes; (void)n_in; (void)out_size; (void)ws_size;
  const float* x     = (const float*)d_in[0];
  const float* ln1s  = (const float*)d_in[1];
  const float* ln1b  = (const float*)d_in[2];
  const float* Wq    = (const float*)d_in[3];
  const float* bq    = (const float*)d_in[4];
  const float* Wk    = (const float*)d_in[5];
  const float* bk    = (const float*)d_in[6];
  const float* Wv    = (const float*)d_in[7];
  const float* bv    = (const float*)d_in[8];
  const float* Wo    = (const float*)d_in[9];
  const float* bo    = (const float*)d_in[10];
  const float* ln2s  = (const float*)d_in[11];
  const float* ln2b  = (const float*)d_in[12];
  const float* W1    = (const float*)d_in[13];
  const float* b1    = (const float*)d_in[14];
  const float* W2    = (const float*)d_in[15];
  const float* b2    = (const float*)d_in[16];
  float* outp = (float*)d_out;

  char* ws = (char*)d_ws;
  const size_t N_ACT = (size_t)MTOT * DMODEL;
  float* xw   = (float*)ws;
  f16*   xn16 = (f16*)(ws + N_ACT * 4);
  f16*   big  = (f16*)(ws + N_ACT * 4 + N_ACT * 2);
  f16*   q16  = big;
  f16*   k16  = big + N_ACT;
  f16*   v16  = big + 2 * N_ACT;
  f16*   h16  = big;
  const size_t WT_ELEMS = (size_t)DMODEL * (3 * DMODEL + DMODEL + DFF + DFF);
  f16* wT = (f16*)(ws + N_ACT * 4 + N_ACT * 2 + (size_t)MTOT * DFF * 2);

  static_assert(SMEM_ATTN_BYTES == 242688, "attention LDS size");
  hipFuncSetAttribute((const void*)attn_wmma,
                      hipFuncAttributeMaxDynamicSharedMemorySize, SMEM_ATTN_BYTES);

  dim3 blk256(256);

  win_split<<<MTOT * (DMODEL / 4) / 256, blk256, 0, stream>>>(x, xw);

  for (int l = 0; l < L_LAYERS; ++l) {
    f16* wTl   = wT + (size_t)l * WT_ELEMS;
    f16* qkvT  = wTl;
    f16* woT   = wTl + (size_t)3 * DMODEL * DMODEL;
    f16* w1T   = woT + (size_t)DMODEL * DMODEL;
    f16* w2T   = w1T + (size_t)DFF * DMODEL;
    size_t wsq = (size_t)l * DMODEL * DMODEL;
    transpose_cvt<<<dim3(DMODEL/32, DMODEL/32), blk256, 0, stream>>>(Wq + wsq, qkvT,                            DMODEL, DMODEL);
    transpose_cvt<<<dim3(DMODEL/32, DMODEL/32), blk256, 0, stream>>>(Wk + wsq, qkvT + (size_t)DMODEL*DMODEL,    DMODEL, DMODEL);
    transpose_cvt<<<dim3(DMODEL/32, DMODEL/32), blk256, 0, stream>>>(Wv + wsq, qkvT + (size_t)2*DMODEL*DMODEL,  DMODEL, DMODEL);
    transpose_cvt<<<dim3(DMODEL/32, DMODEL/32), blk256, 0, stream>>>(Wo + wsq, woT, DMODEL, DMODEL);
    transpose_cvt<<<dim3(DFF/32,   DMODEL/32), blk256, 0, stream>>>(W1 + (size_t)l*DMODEL*DFF, w1T, DMODEL, DFF);
    transpose_cvt<<<dim3(DMODEL/32, DFF/32),   blk256, 0, stream>>>(W2 + (size_t)l*DFF*DMODEL, w2T, DFF, DMODEL);
  }

  for (int l = 0; l < L_LAYERS; ++l) {
    f16* wTl  = wT + (size_t)l * WT_ELEMS;
    f16* wqT  = wTl;
    f16* wkT  = wTl + (size_t)DMODEL * DMODEL;
    f16* wvT  = wTl + (size_t)2 * DMODEL * DMODEL;
    f16* woT  = wTl + (size_t)3 * DMODEL * DMODEL;
    f16* w1T  = woT + (size_t)DMODEL * DMODEL;
    f16* w2T  = w1T + (size_t)DFF * DMODEL;

    ln_f16<<<MTOT, blk256, 0, stream>>>(xw, ln1s + l * DMODEL, ln1b + l * DMODEL, xn16);
    gemm_wmma<<<dim3(MTOT/BM, DMODEL/BN), blk256, 0, stream>>>(
        xn16, DMODEL, wqT, DMODEL, bq + l * DMODEL, q16, nullptr, DMODEL, DMODEL, MODE_F16, 0.125f);
    gemm_wmma<<<dim3(MTOT/BM, DMODEL/BN), blk256, 0, stream>>>(
        xn16, DMODEL, wkT, DMODEL, bk + l * DMODEL, k16, nullptr, DMODEL, DMODEL, MODE_F16, 1.0f);
    gemm_wmma<<<dim3(MTOT/BM, DMODEL/BN), blk256, 0, stream>>>(
        xn16, DMODEL, wvT, DMODEL, bv + l * DMODEL, v16, nullptr, DMODEL, DMODEL, MODE_F16, 1.0f);
    attn_wmma<<<NWIN * NHEADS, blk256, SMEM_ATTN_BYTES, stream>>>(q16, k16, v16, xn16);
    gemm_wmma<<<dim3(MTOT/BM, DMODEL/BN), blk256, 0, stream>>>(
        xn16, DMODEL, woT, DMODEL, bo + l * DMODEL, nullptr, xw, DMODEL, DMODEL, MODE_RES, 1.0f);
    ln_f16<<<MTOT, blk256, 0, stream>>>(xw, ln2s + l * DMODEL, ln2b + l * DMODEL, xn16);
    gemm_wmma<<<dim3(MTOT/BM, DFF/BN), blk256, 0, stream>>>(
        xn16, DMODEL, w1T, DMODEL, b1 + l * DFF, h16, nullptr, DFF, DMODEL, MODE_GELU, 1.0f);
    gemm_wmma<<<dim3(MTOT/BM, DMODEL/BN), blk256, 0, stream>>>(
        h16, DFF, w2T, DFF, b2 + l * DMODEL, nullptr, xw, DMODEL, DFF, MODE_RES, 1.0f);
  }

  win_merge<<<MTOT * (DMODEL / 4) / 256, blk256, 0, stream>>>(xw, outp);
}
